// DiffusionMemory_78623671320763
// MI455X (gfx1250) — matrix operand order repaired and checked by reading, not compiled, not run on hardware
//
#include <hip/hip_runtime.h>

typedef _Float16 half_t;
typedef __attribute__((ext_vector_type(16))) _Float16 v16h;
typedef __attribute__((ext_vector_type(8)))  float    v8f;
typedef unsigned int u32;
typedef u32 v4u __attribute__((ext_vector_type(4)));
typedef int v4i __attribute__((ext_vector_type(4)));
typedef int v8i __attribute__((ext_vector_type(8)));

#define C_DIM 256
#define T_DIM 16
#define P_DIM 1024              // 32*32 spatial positions per frame
#define N_DIM (T_DIM * P_DIM)   // 16384
#define NH 8
#define HD 32
#define KV_CHUNK 256            // keys per TDM-staged LDS chunk

// ---------------------------------------------------------------------------
// WMMA fragment helpers (wave32; lane half h holds k = 16h..16h+15 in both A and B)
// ---------------------------------------------------------------------------
__device__ __forceinline__ int lane_id() { return threadIdx.x & 31; }

// A fragment 16x32 f16 from row-major src (ld elements per row).
__device__ __forceinline__ v16h load_a(const half_t* __restrict__ src, int ld,
                                       int r0, int k0) {
    const int l = lane_id(), lm = l & 15, hs = (l >> 4) & 1;
    const half_t* p = src + (size_t)(r0 + lm) * ld + k0 + hs * 16;
    v16h a;
#pragma unroll
    for (int h = 0; h < 16; ++h) a[h] = p[h];
    return a;
}

// B fragment 32x16 f16 where B[k][n] = W[n0+n][k0+k] (contraction contiguous)
__device__ __forceinline__ v16h load_b_rm(const half_t* __restrict__ W, int ld,
                                          int n0, int k0) {
    const int l = lane_id(), lm = l & 15, hs = (l >> 4) & 1;
    const half_t* p = W + (size_t)(n0 + lm) * ld + k0 + hs * 16;
    v16h b;
#pragma unroll
    for (int h = 0; h < 16; ++h) b[h] = p[h];
    return b;
}

// B fragment 32x16 f16 where B[k][n] = V[k0+k][n0+n] (strided K)
__device__ __forceinline__ v16h load_b_cm(const half_t* __restrict__ V, int ld,
                                          int k0, int n0) {
    const int l = lane_id(), lm = l & 15, hs = (l >> 4) & 1;
    const half_t* p = V + (size_t)(k0 + hs * 16) * ld + n0 + lm;
    v16h b;
#pragma unroll
    for (int h = 0; h < 16; ++h) b[h] = p[h * ld];
    return b;
}

__device__ __forceinline__ v8f wmma16(v16h a, v16h b, v8f c) {
    return __builtin_amdgcn_wmma_f32_16x16x32_f16(false, a, false, b,
                                                  (short)0, c, false, false);
}

// ---------------------------------------------------------------------------
// TDM: DMA a 2-D f16 tile [rows x row_elems] (row-major, stride row_elems)
// from global memory into LDS.  D# layout per CDNA5 ISA 8.3/8.4.
// ---------------------------------------------------------------------------
__device__ __forceinline__ void tdm_load_2d_f16(u32 lds_byte_addr,
                                                const void* gaddr,
                                                u32 rows, u32 row_elems) {
    unsigned long long ga = (unsigned long long)gaddr;
    v4u g0;
    g0[0] = 1u;                                        // count=1, no gather
    g0[1] = lds_byte_addr;                             // lds_addr
    g0[2] = (u32)(ga & 0xFFFFFFFFu);                   // global_addr[31:0]
    g0[3] = (u32)((ga >> 32) & 0x1FFFFFFu) | (2u << 30); // addr[56:32] | type=2
    v8i g1;
    g1[0] = (int)(1u << 16);                           // data_size=1 (2 bytes)
    g1[1] = (int)((row_elems & 0xFFFFu) << 16);        // tensor_dim0 lo
    g1[2] = (int)((rows & 0xFFFFu) << 16);             // dim0 hi=0 | tensor_dim1 lo
    g1[3] = (int)((row_elems & 0xFFFFu) << 16);        // dim1 hi=0 | tile_dim0
    g1[4] = (int)(rows & 0xFFFFu);                     // tile_dim1 (tile_dim2=0)
    g1[5] = (int)row_elems;                            // tensor_dim0_stride lo
    g1[6] = 0;
    g1[7] = 0;
    v4i z4 = {0, 0, 0, 0};
#if __clang_major__ >= 23
    v8i z8 = {0, 0, 0, 0, 0, 0, 0, 0};
    __builtin_amdgcn_tensor_load_to_lds(g0, g1, z4, z4, z8, 0);
#else
    __builtin_amdgcn_tensor_load_to_lds(g0, g1, z4, z4, 0);
#endif
}

// ---------------------------------------------------------------------------
// K1: x [c][n] f32 -> x_h [n][c] f16 and xp_h = (x + pos_emb) [n][c] f16
// ---------------------------------------------------------------------------
__global__ void prep_x_kernel(const float* __restrict__ x,
                              const float* __restrict__ pos,
                              half_t* __restrict__ x_h,
                              half_t* __restrict__ xp_h) {
    int idx = blockIdx.x * blockDim.x + threadIdx.x;
    if (idx >= N_DIM * C_DIM) return;
    int c = idx >> 14;
    int n = idx & (N_DIM - 1);
    float v = x[idx];
    int t = n >> 10;
    float pe = pos[c * T_DIM + t];
    x_h[(size_t)n * C_DIM + c]  = (half_t)v;
    xp_h[(size_t)n * C_DIM + c] = (half_t)(v + pe);
}

// ---------------------------------------------------------------------------
// K2: weights -> f16.  tc_w [o][c][3] split into 3 matrices wt[dt][o][c]
// ---------------------------------------------------------------------------
__global__ void prep_w_kernel(const float* __restrict__ tcw,
                              const float* __restrict__ qw,
                              const float* __restrict__ kw,
                              const float* __restrict__ vw,
                              const float* __restrict__ ow,
                              half_t* __restrict__ wt,
                              half_t* __restrict__ wq, half_t* __restrict__ wk,
                              half_t* __restrict__ wv, half_t* __restrict__ wo) {
    int idx = blockIdx.x * blockDim.x + threadIdx.x;
    if (idx >= C_DIM * C_DIM) return;
    wq[idx] = (half_t)qw[idx];
    wk[idx] = (half_t)kw[idx];
    wv[idx] = (half_t)vw[idx];
    wo[idx] = (half_t)ow[idx];
#pragma unroll
    for (int dt = 0; dt < 3; ++dt)
        wt[dt * (C_DIM * C_DIM) + idx] = (half_t)tcw[idx * 3 + dt];
}

// ---------------------------------------------------------------------------
// K3: Q/K/V projections.  One wave per 16-row tile; A-frags reused 48x.
// ---------------------------------------------------------------------------
__global__ __launch_bounds__(256)
void qkv_kernel(const half_t* __restrict__ xp,
                const half_t* __restrict__ wq, const half_t* __restrict__ wk,
                const half_t* __restrict__ wv,
                const float* __restrict__ qb, const float* __restrict__ kb,
                const float* __restrict__ vb,
                half_t* __restrict__ qg, half_t* __restrict__ kg,
                half_t* __restrict__ vg) {
    const int wave = threadIdx.x >> 5;
    const int rt = blockIdx.x * 8 + wave;
    const int r0 = rt * 16;
    const int l = lane_id(), lm = l & 15, hs = (l >> 4) & 1;

    v16h A[8];
#pragma unroll
    for (int kc = 0; kc < 8; ++kc) A[kc] = load_a(xp, C_DIM, r0, kc * 32);

    const half_t* Ws[3] = {wq, wk, wv};
    const float*  Bs[3] = {qb, kb, vb};
    half_t*       Og[3] = {qg, kg, vg};
    const float scale_q = 0.17677669529663687f;   // 32^-0.5

#pragma unroll
    for (int out = 0; out < 3; ++out) {
        const float scl = (out == 0) ? scale_q : 1.0f;
        for (int ct = 0; ct < 16; ++ct) {
            v8f acc = {};
#pragma unroll
            for (int kc = 0; kc < 8; ++kc)
                acc = wmma16(A[kc], load_b_rm(Ws[out], C_DIM, ct * 16, kc * 32), acc);
            const int co = ct * 16 + lm;
            const float bias = Bs[out][co];
            const int head = co >> 5, d = co & 31;
#pragma unroll
            for (int r = 0; r < 8; ++r) {
                int n = r0 + hs * 8 + r;
                int t = n >> 10, p = n & (P_DIM - 1);
                Og[out][((size_t)(head * T_DIM + t) * P_DIM + p) * HD + d] =
                    (half_t)((acc[r] + bias) * scl);
            }
        }
    }
}

// ---------------------------------------------------------------------------
// K4: attention.  One workgroup per (head, t, 128-q-row slice) -> 1024 WGs.
// Each wave owns one 16-row q-tile (online-softmax state in registers).
// K/V staged into LDS in 256-key chunks by the Tensor Data Mover; wave 0
// issues the DMA, waits on TENSORcnt, then the workgroup barrier releases
// all 8 waves to consume the tiles.
// ---------------------------------------------------------------------------
__global__ __launch_bounds__(256)
void attn_kernel(const half_t* __restrict__ qg, const half_t* __restrict__ kg,
                 const half_t* __restrict__ vg, half_t* __restrict__ ao) {
    __shared__ half_t Kt[KV_CHUNK * HD];      // 16 KB
    __shared__ half_t Vt[KV_CHUNK * HD];      // 16 KB
    __shared__ half_t p_lds[8][16][40];       // 10 KB P-tile relayout scratch

    const int wave = threadIdx.x >> 5;
    const int ht = blockIdx.x >> 3;           // head*16 + t
    const int qc = blockIdx.x & 7;            // which 128-row q slice
    const int head = ht >> 4, t = ht & 15;
    const half_t* Q = qg + (size_t)ht * P_DIM * HD;
    const half_t* K = kg + (size_t)ht * P_DIM * HD;
    const half_t* V = vg + (size_t)ht * P_DIM * HD;
    const int l = lane_id(), lm = l & 15, hs = (l >> 4) & 1;

    const int p0 = qc * 128 + wave * 16;      // this wave's q-tile base row
    v16h Aq = load_a(Q, HD, p0, 0);           // 16x32 Q tile (scale pre-folded)

    float m[8], lden[8];
    v8f olo = {}, ohi = {};
#pragma unroll
    for (int r = 0; r < 8; ++r) { m[r] = -1e30f; lden[r] = 0.0f; }

    for (int chunk = 0; chunk < 4; ++chunk) {
        __syncthreads();                      // previous chunk fully consumed
        if (wave == 0) {
            tdm_load_2d_f16((u32)(uintptr_t)Kt, K + (size_t)chunk * KV_CHUNK * HD,
                            KV_CHUNK, HD);
            tdm_load_2d_f16((u32)(uintptr_t)Vt, V + (size_t)chunk * KV_CHUNK * HD,
                            KV_CHUNK, HD);
            __builtin_amdgcn_s_wait_tensorcnt(0);
        }
        __syncthreads();                      // LDS tiles visible to all waves

        for (int kt = 0; kt < 8; ++kt) {      // 8 x 32 keys per chunk
            const int pq0 = kt * 32;          // chunk-local key row
            v8f z0 = {}, z1 = {};
            v8f s0 = wmma16(Aq, load_b_rm(Kt, HD, pq0, 0),      z0);
            v8f s1 = wmma16(Aq, load_b_rm(Kt, HD, pq0 + 16, 0), z1);
#pragma unroll
            for (int r = 0; r < 8; ++r) {
                float tmax = fmaxf(s0[r], s1[r]);
#pragma unroll
                for (int off = 1; off < 16; off <<= 1)
                    tmax = fmaxf(tmax, __shfl_xor(tmax, off, 32));
                float nm   = fmaxf(m[r], tmax);
                float corr = __expf(m[r] - nm);
                float pa   = __expf(s0[r] - nm);
                float pb   = __expf(s1[r] - nm);
                float rs = pa + pb;
#pragma unroll
                for (int off = 1; off < 16; off <<= 1)
                    rs += __shfl_xor(rs, off, 32);
                lden[r] = lden[r] * corr + rs;
                m[r] = nm;
                olo[r] *= corr; ohi[r] *= corr;
                p_lds[wave][hs * 8 + r][lm]      = (half_t)pa;
                p_lds[wave][hs * 8 + r][16 + lm] = (half_t)pb;
            }
            v16h Ap = load_a(&p_lds[wave][0][0], 40, 0, 0);
            olo = wmma16(Ap, load_b_cm(Vt, HD, pq0, 0),  olo);
            ohi = wmma16(Ap, load_b_cm(Vt, HD, pq0, 16), ohi);
        }
    }

#pragma unroll
    for (int r = 0; r < 8; ++r) {
        int n = t * P_DIM + p0 + hs * 8 + r;
        float inv = 1.0f / lden[r];
        ao[(size_t)n * C_DIM + head * HD + lm]      = (half_t)(olo[r] * inv);
        ao[(size_t)n * C_DIM + head * HD + 16 + lm] = (half_t)(ohi[r] * inv);
    }
}

// ---------------------------------------------------------------------------
// K5: fused temporal conv (3 shifted GEMMs on x_h) + O-projection + biases.
// ---------------------------------------------------------------------------
__global__ __launch_bounds__(256)
void final_kernel(const half_t* __restrict__ xh, const half_t* __restrict__ aoh,
                  const half_t* __restrict__ wt, const half_t* __restrict__ wo,
                  const float* __restrict__ tcb, const float* __restrict__ ob,
                  float* __restrict__ out) {
    const int wave = threadIdx.x >> 5;
    const int rt = blockIdx.x * 8 + wave;
    const int r0 = rt * 16;
    const int t = r0 >> 10;
    const int l = lane_id(), lm = l & 15, hs = (l >> 4) & 1;

    for (int ct = 0; ct < 16; ++ct) {
        v8f acc = {};
#pragma unroll
        for (int dt = 0; dt < 3; ++dt) {
            int tt = t + dt - 1;
            if (tt < 0 || tt >= T_DIM) continue;          // zero padding
            const half_t* wdt = wt + dt * (C_DIM * C_DIM);
            const int rs = r0 + (dt - 1) * P_DIM;
#pragma unroll
            for (int kc = 0; kc < 8; ++kc)
                acc = wmma16(load_a(xh, C_DIM, rs, kc * 32),
                             load_b_rm(wdt, C_DIM, ct * 16, kc * 32), acc);
        }
#pragma unroll
        for (int kc = 0; kc < 8; ++kc)
            acc = wmma16(load_a(aoh, C_DIM, r0, kc * 32),
                         load_b_rm(wo, C_DIM, ct * 16, kc * 32), acc);
        const int o = ct * 16 + lm;
        const float bias = tcb[o] + ob[o];
#pragma unroll
        for (int r = 0; r < 8; ++r) {
            int n = r0 + hs * 8 + r;
            out[(size_t)o * N_DIM + n] = acc[r] + bias;
        }
    }
}

// ---------------------------------------------------------------------------
// Launcher
// ---------------------------------------------------------------------------
extern "C" void kernel_launch(void* const* d_in, const int* in_sizes, int n_in,
                              void* d_out, int out_size, void* d_ws, size_t ws_size,
                              hipStream_t stream) {
    const float* x    = (const float*)d_in[0];
    const float* tc_w = (const float*)d_in[1];
    const float* tc_b = (const float*)d_in[2];
    const float* q_w  = (const float*)d_in[3];
    const float* q_b  = (const float*)d_in[4];
    const float* k_w  = (const float*)d_in[5];
    const float* k_b  = (const float*)d_in[6];
    const float* v_w  = (const float*)d_in[7];
    const float* v_b  = (const float*)d_in[8];
    const float* o_w  = (const float*)d_in[9];
    const float* o_b  = (const float*)d_in[10];
    const float* pos  = (const float*)d_in[11];

    char* ws = (char*)d_ws;
    size_t off = 0;
    auto alloc = [&](size_t bytes) -> void* {
        void* p = ws + off;
        off += (bytes + 255) & ~(size_t)255;
        return p;
    };
    const size_t ACT = (size_t)N_DIM * C_DIM * sizeof(half_t);   // 8 MB
    const size_t WMT = (size_t)C_DIM * C_DIM * sizeof(half_t);   // 128 KB

    half_t* x_h  = (half_t*)alloc(ACT);
    half_t* xp_h = (half_t*)alloc(ACT);
    half_t* q_h  = (half_t*)alloc(ACT);
    half_t* k_h  = (half_t*)alloc(ACT);
    half_t* v_h  = (half_t*)alloc(ACT);
    half_t* ao_h = (half_t*)alloc(ACT);
    half_t* wq_h = (half_t*)alloc(WMT);
    half_t* wk_h = (half_t*)alloc(WMT);
    half_t* wv_h = (half_t*)alloc(WMT);
    half_t* wo_h = (half_t*)alloc(WMT);
    half_t* wt_h = (half_t*)alloc(3 * WMT);

    prep_x_kernel<<<(N_DIM * C_DIM + 255) / 256, 256, 0, stream>>>(x, pos, x_h, xp_h);
    prep_w_kernel<<<(C_DIM * C_DIM + 255) / 256, 256, 0, stream>>>(
        tc_w, q_w, k_w, v_w, o_w, wt_h, wq_h, wk_h, wv_h, wo_h);
    qkv_kernel<<<128, 256, 0, stream>>>(xp_h, wq_h, wk_h, wv_h,
                                        q_b, k_b, v_b, q_h, k_h, v_h);
    attn_kernel<<<1024, 256, 0, stream>>>(q_h, k_h, v_h, ao_h);
    final_kernel<<<128, 256, 0, stream>>>(x_h, ao_h, wt_h, wo_h,
                                          tc_b, o_b, (float*)d_out);
}
